// FractalDavid_38319698215362
// MI455X (gfx1250) — compile-verified, matrix operand order checked by reading, not run on hardware
//
#include <hip/hip_runtime.h>
#include <hip/hip_fp16.h>
#include <cstdint>

// ---------------------------------------------------------------------------
// MI455X (gfx1250): WMMA f16 GEMM pipeline with async global->LDS staging.
//   - v_wmma_f32_16x16x32_f16, 8 waves/block, BM=64 x BN=128 x BK=32 tiles
//   - f16 operand tiles staged via GLOBAL_LOAD_ASYNC_TO_LDS_B128 (ASYNCcnt)
//   - f32 operand tiles converted in-register + ds_store_b128, with
//     global_prefetch_b8 on the next K tile
// ---------------------------------------------------------------------------

typedef __attribute__((ext_vector_type(16))) _Float16 v16h;
typedef __attribute__((ext_vector_type(8)))  _Float16 v8h;
typedef __attribute__((ext_vector_type(8)))  float    v8f;
typedef __attribute__((ext_vector_type(4)))  float    v4f;

union V16H { v16h v; v8h h[2]; };

__device__ __forceinline__ v8f wmma_f16(v16h a, v16h b, v8f c) {
    // (neg_a, A, neg_b, B, c_mod, C, reuse_a, reuse_b)
    return __builtin_amdgcn_wmma_f32_16x16x32_f16(false, a, false, b,
                                                  (short)0, c, false, false);
}

__device__ __forceinline__ float warp_sum32(float v) {
    #pragma unroll
    for (int m = 16; m >= 1; m >>= 1) v += __shfl_xor(v, m, 32);
    return v;
}

__device__ __forceinline__ void wait_async_lds() {
    asm volatile("s_wait_asynccnt 0x0" ::: "memory");
}

// ---------------------------------------------------------------------------
// Tile loaders into LDS (ROWS x 32 halves, row stride 32).
// ---------------------------------------------------------------------------

// f16 source: per-lane async DMA, 16B chunk, no VGPR staging.
template <int ROWS>
__device__ __forceinline__ void load_tile_f16_async(
    _Float16* __restrict__ lds, const _Float16* __restrict__ src, int ld,
    int rowBase, int k0, int nrows) {
    const uint32_t base = (uint32_t)(uintptr_t)lds;   // LDS byte offset
    #pragma unroll
    for (int j = 0; j < ROWS / 64; ++j) {
        const int lin = threadIdx.x + j * 256;
        const int r   = lin >> 2;
        const int c   = (lin & 3) * 8;
        const int gr  = rowBase + r;
        if (gr < nrows) {
            const uint32_t la = base + (uint32_t)(r * 32 + c) * 2u;
            const uint64_t ga = (uint64_t)(src + (size_t)gr * ld + k0 + c);
            asm volatile("global_load_async_to_lds_b128 %0, %1, off"
                         :: "v"(la), "v"(ga) : "memory");
        } else {
            v8h z = {};
            *(v8h*)(lds + r * 32 + c) = z;            // pad tail rows (N=1000)
        }
    }
}

// f32 source: load 2x b128, convert to f16, ds_store_b128; prefetch next tile.
template <int ROWS>
__device__ __forceinline__ void load_tile_f32_cvt(
    _Float16* __restrict__ lds, const float* __restrict__ src, int ld,
    int rowBase, int k0, int K, int nrows) {
    #pragma unroll
    for (int j = 0; j < ROWS / 64; ++j) {
        const int lin = threadIdx.x + j * 256;
        const int r   = lin >> 2;
        const int c   = (lin & 3) * 8;
        const int gr  = rowBase + r;
        v8h val = {};
        if (gr < nrows) {
            const float* p = src + (size_t)gr * ld + k0 + c;
            v4f lo = *(const v4f*)p;
            v4f hi = *(const v4f*)(p + 4);
            #pragma unroll
            for (int q = 0; q < 4; ++q) {
                val[q]     = (_Float16)lo[q];
                val[4 + q] = (_Float16)hi[q];
            }
            if (k0 + 32 < K) __builtin_prefetch(p + 32, 0, 0);  // next K tile
        }
        *(v8h*)(lds + r * 32 + c) = val;
    }
}

// Load one 16x32 f16 WMMA fragment from LDS (row stride 32 halves).
// ISA layout: lanes 0-15 hold M=lane, K {0..7,16..23}; lanes 16-31 hold
// M=lane-16, K {8..15,24..31}.  Two ds_load_b128 per lane.
__device__ __forceinline__ v16h frag_ld(const _Float16* __restrict__ lds,
                                        int tileRow) {
    const int lane = threadIdx.x & 31;
    const int row  = tileRow + (lane & 15);
    const int kb   = (lane >> 4) * 8;
    V16H f;
    f.h[0] = *(const v8h*)(lds + row * 32 + kb);
    f.h[1] = *(const v8h*)(lds + row * 32 + kb + 16);
    return f.v;
}

// Cantor staircase for pos=idx, max_pos=5 (matches reference _cantor_stairs).
__device__ __forceinline__ float cantor_val(int idx, float alpha) {
    float x = (float)idx * 0.25f;
    x = fminf(fmaxf(x, 1e-6f), 1.f - 1e-6f);
    float Cx = 0.f, w = 0.5f;
    const float inv_tau = 1.f / (0.25f + 1e-8f);
    #pragma unroll
    for (int l = 0; l < 12; ++l) {
        float y = x * 3.f;
        float d0 = y - 0.5f, d1 = y - 1.5f, d2 = y - 2.5f;
        float e0 = __expf(-d0 * d0 * inv_tau);
        float e1 = __expf(-d1 * d1 * inv_tau);
        float e2 = __expf(-d2 * d2 * inv_tau);
        float inv = 1.f / (e0 + e1 + e2);
        Cx += (e2 + alpha * e1) * inv * w;
        x = y - floorf(y);
        w *= 0.5f;
    }
    return fminf(fmaxf(Cx, 0.f), 1.f);
}

// ---------------------------------------------------------------------------
// WMMA GEMM: C[M,N] = epi(A[M,K] @ B[N,K]^T).  Block 64x128, BK=32.
// Wave wv: M-tile (wv>>1)*16, four N-tiles at (wv&1)*64 .. +48.
// EPI 0: f32 store.  EPI 1: relu(x + bias[n]) -> f16 store.
// ---------------------------------------------------------------------------
template <typename TA, int EPI>
__global__ __launch_bounds__(256)
void gemm_wmma(const TA* __restrict__ A, int lda,
               const float* __restrict__ B, int ldb,
               const float* __restrict__ bias,
               void* __restrict__ C, int ldc, int M, int N, int K) {
    __shared__ _Float16 As[64 * 32];
    __shared__ _Float16 Bs[128 * 32];
    const int bm = blockIdx.x * 64;
    const int bn = blockIdx.y * 128;
    const int wv = threadIdx.x >> 5;
    const int ar = (wv >> 1) * 16;
    const int bc = (wv & 1) * 64;

    v8f acc[4] = {};
    for (int k0 = 0; k0 < K; k0 += 32) {
        if constexpr (sizeof(TA) == 2)
            load_tile_f16_async<64>(As, (const _Float16*)A, lda, bm, k0, M);
        else
            load_tile_f32_cvt<64>(As, (const float*)A, lda, bm, k0, K, M);
        load_tile_f32_cvt<128>(Bs, B, ldb, bn, k0, K, N);
        if constexpr (sizeof(TA) == 2) wait_async_lds();
        __syncthreads();

        const v16h a = frag_ld(As, ar);
        #pragma unroll
        for (int t = 0; t < 4; ++t) {
            v16h b = frag_ld(Bs, bc + 16 * t);
            acc[t] = wmma_f16(a, b, acc[t]);
        }
        __syncthreads();
    }

    const int lane = threadIdx.x & 31;
    const int mlo  = (lane >> 4) * 8;      // lanes 16-31 -> M += 8
    const int nl   = lane & 15;
    #pragma unroll
    for (int r = 0; r < 8; ++r) {
        const int row = bm + ar + mlo + r;
        if (row >= M) continue;
        #pragma unroll
        for (int t = 0; t < 4; ++t) {
            const int col = bn + bc + 16 * t + nl;
            if (col >= N) continue;
            if constexpr (EPI == 0) {
                ((float*)C)[(size_t)row * ldc + col] = acc[t][r];
            } else {
                ((_Float16*)C)[(size_t)row * ldc + col] =
                    (_Float16)fmaxf(acc[t][r] + bias[col], 0.f);
            }
        }
    }
}

// ---------------------------------------------------------------------------
// Fused head: feat = Z@Pn^T, cry = Z@An^T (shared A fragment, 8 WMMA/K-step),
// then the log-domain epilogue, accumulating fw[i]*logits into out (2048x1000).
// All operands f16 -> fully async-staged.
// ---------------------------------------------------------------------------
__global__ __launch_bounds__(256)
void head_logits_wmma(const _Float16* __restrict__ Z,
                      const _Float16* __restrict__ Pn,
                      const _Float16* __restrict__ An,
                      const float* __restrict__ alpha_p,
                      const float* __restrict__ fusion_w,   // 5
                      const float* __restrict__ geo_w,      // 3
                      float* __restrict__ out, int scaleIdx, int K) {
    __shared__ _Float16 As[64 * 32];
    __shared__ _Float16 B1s[128 * 32];
    __shared__ _Float16 B2s[128 * 32];
    const int bm = blockIdx.x * 64;
    const int bn = blockIdx.y * 128;
    const int wv = threadIdx.x >> 5;
    const int ar = (wv >> 1) * 16;
    const int bc = (wv & 1) * 64;

    v8f aF[4] = {}, aC[4] = {};
    for (int k0 = 0; k0 < K; k0 += 32) {
        load_tile_f16_async<64>(As, Z, K, bm, k0, 2048);
        load_tile_f16_async<128>(B1s, Pn, K, bn, k0, 1000);
        load_tile_f16_async<128>(B2s, An, K, bn, k0, 1000);
        wait_async_lds();
        __syncthreads();

        const v16h a = frag_ld(As, ar);
        #pragma unroll
        for (int t = 0; t < 4; ++t) {
            v16h bp = frag_ld(B1s, bc + 16 * t);
            aF[t] = wmma_f16(a, bp, aF[t]);
            v16h ba = frag_ld(B2s, bc + 16 * t);
            aC[t] = wmma_f16(a, ba, aC[t]);
        }
        __syncthreads();
    }

    // Per-thread scalar epilogue constants (tiny: 3/5-wide softmaxes).
    const float alpha = alpha_p[0];
    float g0 = geo_w[0], g1 = geo_w[1], g2 = geo_w[2];
    float gm = fmaxf(g0, fmaxf(g1, g2));
    float e0 = __expf(g0 - gm), e1 = __expf(g1 - gm), e2 = __expf(g2 - gm);
    float gs = 1.f / (e0 + e1 + e2);
    const float w0 = e0 * gs, w1 = e1 * gs, w2 = e2 * gs;
    float fmx = fusion_w[0];
    #pragma unroll
    for (int j = 1; j < 5; ++j) fmx = fmaxf(fmx, fusion_w[j]);
    float fs = 0.f, fe[5];
    #pragma unroll
    for (int j = 0; j < 5; ++j) { fe[j] = __expf(fusion_w[j] - fmx); fs += fe[j]; }
    const float fwI = fe[scaleIdx] / fs;
    const float cv  = cantor_val(scaleIdx, alpha);

    const int lane = threadIdx.x & 31;
    const int mlo  = (lane >> 4) * 8;
    const int nl   = lane & 15;

    auto emit = [&](int row, float f, float c, int col) {
        if (col >= 1000) return;
        float featc = fminf(fmaxf((f + 1.f) * 0.5f, 1e-8f), 1.f);
        float cryc  = fminf(fmaxf((c + 1.f) * 0.5f, 1e-8f), 1.f);
        float d     = cv - (float)col * (1.f / 999.f);
        float cantc = fminf(fmaxf(__expf(-d * d * 10.f), 1e-8f), 1.f);
        // compat^clip in log domain: t = sum w_i*log(c_i), clip to [log EPS, 0]
        float t = w0 * __logf(featc) + w1 * __logf(cantc) + w2 * __logf(cryc);
        t = fminf(fmaxf(t, -18.4206807f), 0.f);
        out[(size_t)row * 1000 + col] += fwI * t * (1.f / 0.07f);
    };

    #pragma unroll
    for (int r = 0; r < 8; ++r) {
        const int row = bm + ar + mlo + r;
        #pragma unroll
        for (int t = 0; t < 4; ++t)
            emit(row, aF[t][r], aC[t][r], bn + bc + 16 * t + nl);
    }
}

// Row L2-normalize (f32 in, stride inStride) -> packed f16 out (stride len).
// One wave32 per row, 8 rows per block.
__global__ __launch_bounds__(256)
void normalize_rows(const float* __restrict__ in, int inStride,
                    _Float16* __restrict__ outp, int len, int nrows) {
    const int row = blockIdx.x * 8 + (threadIdx.x >> 5);
    if (row >= nrows) return;
    const int lane = threadIdx.x & 31;
    const float* src = in + (size_t)row * inStride;
    float ss = 0.f;
    for (int i = lane; i < len; i += 32) { float v = src[i]; ss += v * v; }
    ss = warp_sum32(ss);
    const float sc = 1.f / fmaxf(sqrtf(ss), 1e-12f);
    _Float16* dst = outp + (size_t)row * len;
    for (int i = lane; i < len; i += 32) dst[i] = (_Float16)(src[i] * sc);
}

__global__ __launch_bounds__(256)
void zero_f32(float* __restrict__ p, int n) {
    int i = blockIdx.x * 256 + threadIdx.x;
    if (i < n) p[i] = 0.f;
}

// ---------------------------------------------------------------------------
extern "C" void kernel_launch(void* const* d_in, const int* in_sizes, int n_in,
                              void* d_out, int out_size, void* d_ws, size_t ws_size,
                              hipStream_t stream) {
    (void)in_sizes; (void)n_in; (void)ws_size;
    static const int SC[5] = {384, 512, 768, 1024, 1280};

    const float* features = (const float*)d_in[0];
    const float* crystals[5];
    for (int i = 0; i < 5; ++i) crystals[i] = (const float*)d_in[1 + i];
    const float* alpha    = (const float*)d_in[6];
    const float* fusion_w = (const float*)d_in[7];
    // params[f's{s}'] in insertion order: w1, b1, w2, cantor_protos,
    // class_protos, geo_weights  ->  base = 8 + 6*i
    const int PB = 8;

    // Workspace layout (buffers reused across the 5 sequential scales).
    char* ws = (char*)d_ws;
    _Float16* H  = (_Float16*)ws;                               // 2048 x 2560 f16
    float*    Zr = (float*)(ws + 2048ull * 2560 * 2);           // 2048 x 1280 f32
    _Float16* Zh = (_Float16*)((char*)Zr + 2048ull * 1280 * 4); // 2048 x 1280 f16
    _Float16* Pn = (_Float16*)((char*)Zh + 2048ull * 1280 * 2); // 1000 x 1280 f16
    _Float16* An = (_Float16*)((char*)Pn + 1000ull * 1280 * 2); // 1000 x 1280 f16

    float* outp = (float*)d_out;
    zero_f32<<<(out_size + 255) / 256, 256, 0, stream>>>(outp, out_size);

    for (int i = 0; i < 5; ++i) {
        const int s = SC[i], belly = 2 * s;
        const float* w1  = (const float*)d_in[PB + 6 * i + 0];  // (belly,768)
        const float* b1  = (const float*)d_in[PB + 6 * i + 1];  // (belly,)
        const float* w2  = (const float*)d_in[PB + 6 * i + 2];  // (s,belly)
        const float* cp  = (const float*)d_in[PB + 6 * i + 4];  // (1000,s)
        const float* geo = (const float*)d_in[PB + 6 * i + 5];  // (3,)

        // Normalized class protos and crystal anchors -> f16.
        normalize_rows<<<125, 256, 0, stream>>>(cp, s, Pn, s, 1000);
        normalize_rows<<<125, 256, 0, stream>>>(crystals[i], 5 * s, An, s, 1000);

        // H = relu(X @ W1^T + b1) -> f16
        gemm_wmma<float, 1><<<dim3(2048 / 64, belly / 128), 256, 0, stream>>>(
            features, 768, w1, 768, b1, (void*)H, belly, 2048, belly, 768);

        // Zraw = H @ W2^T -> f32
        gemm_wmma<_Float16, 0><<<dim3(2048 / 64, s / 128), 256, 0, stream>>>(
            H, belly, w2, belly, nullptr, (void*)Zr, s, 2048, s, belly);

        // Z = normalize_rows(Zraw) -> f16
        normalize_rows<<<256, 256, 0, stream>>>(Zr, s, Zh, s, 2048);

        // Fused dual GEMM + epilogue, accumulate fw[i]*logits into out.
        head_logits_wmma<<<dim3(2048 / 64, 8), 256, 0, stream>>>(
            Zh, Pn, An, alpha, fusion_w, geo, outp, i, s);
    }
}
